// ConvSelfAttention_22539988369994
// MI455X (gfx1250) — matrix operand order repaired and checked by reading, not compiled, not run on hardware
//
#include <hip/hip_runtime.h>

// ---------------------------------------------------------------------------
// ConvSelfAttention for MI455X (gfx1250, wave32, WMMA bf16 16x16x32)
//   B=8, C=512, C8=64, N=2048
//   Pipeline:
//     proj_kernel x3  : f = Wq*x, g = Wk*x, h = Wv*x   (bf16 in workspace)
//     attn_kernel     : flash-attention over key axis n, fused gamma*O/sum + x
//   gfx1250 features: v_wmma_f32_16x16x32_bf16, global_load_async_to_lds_b128
//   (ASYNCcnt-tracked staging), wave32 WGP-mode block sizing.
//   S/P score tiles are stored m-major in LDS so softmax and the PV
//   B-fragment loads are contiguous (ds_load_b128-mergeable).
// ---------------------------------------------------------------------------

#define BB 8
#define CC 512
#define C8V 64
#define NN 2048

typedef __attribute__((ext_vector_type(16))) __bf16 v16bf;
typedef __attribute__((ext_vector_type(8)))  float  v8f;
typedef int v4i_vs __attribute__((vector_size(16)));   // matches builtin pointee

// ---- gfx1250 async copy: global -> LDS, 16B per lane, tracked by ASYNCcnt --
__device__ __forceinline__ void async_copy_b128(const void* gsrc, void* ldst) {
#if __has_builtin(__builtin_amdgcn_global_load_async_to_lds_b128)
  __builtin_amdgcn_global_load_async_to_lds_b128(
      (__attribute__((address_space(1))) v4i_vs*)gsrc,
      (__attribute__((address_space(3))) v4i_vs*)ldst,
      /*offset=*/0, /*cpol=*/0);
#else
  const unsigned lds_off =
      (unsigned)(size_t)(__attribute__((address_space(3))) void*)ldst;
  asm volatile("global_load_async_to_lds_b128 %0, %1, off"
               :: "v"(lds_off), "v"(gsrc) : "memory");
#endif
}

__device__ __forceinline__ void wait_async0() {
#if __has_builtin(__builtin_amdgcn_s_wait_asynccnt)
  __builtin_amdgcn_s_wait_asynccnt(0);
#else
  asm volatile("s_wait_asynccnt 0x0" ::: "memory");
#endif
}

// D = A(16x32 bf16) * B(32x16 bf16) + C(16x16 f32)
__device__ __forceinline__ v8f wmma_bf16(v16bf a, v16bf b, v8f c) {
  return __builtin_amdgcn_wmma_f32_16x16x32_bf16(
      /*neg_a=*/false, a, /*neg_b=*/false, b,
      /*c_mod=*/(short)0, c, /*reuse_a=*/false, /*reuse_b=*/false);
}

// A-fragment loader (16x32, bf16).  Element i of lane L:
//   M = m_base + (L&15);  K = k_base + ((L>>4)&1)*16 + i
__device__ __forceinline__ v16bf load_frag_a(const __bf16* s, int m_base, int k_base,
                                             int m_stride, int k_stride, int lane) {
  const int m = m_base + (lane & 15);
  const int half = (lane >> 4) & 1;
  v16bf a;
#pragma unroll
  for (int i = 0; i < 16; ++i) {
    const int k = k_base + half * 16 + i;
    a[i] = s[m * m_stride + k * k_stride];
  }
  return a;
}

// B-fragment loader (32x16, bf16).  Element i of lane L:
//   N = n_base + (L&15);  K = k_base + ((L>>4)&1)*16 + i
__device__ __forceinline__ v16bf load_frag_b(const __bf16* s, int k_base, int n_base,
                                             int k_stride, int n_stride, int lane) {
  const int n = n_base + (lane & 15);
  const int half = (lane >> 4) & 1;
  v16bf b;
#pragma unroll
  for (int i = 0; i < 16; ++i) {
    const int k = k_base + half * 16 + i;
    b[i] = s[k * k_stride + n * n_stride];
  }
  return b;
}

// ---------------------------------------------------------------------------
// Projection GEMM:  out[b, o, n] = sum_c W[o, c] * x[b, c, n], stored as bf16.
// Workgroup: 128 threads (4 waves), tile 64(o) x 64(n), K-step 32.
// ---------------------------------------------------------------------------
__global__ __launch_bounds__(128) void proj_kernel(
    const float* __restrict__ x, const float* __restrict__ W,
    __bf16* __restrict__ out, int OutC) {
  __shared__ __bf16 Ws[64 * 32];   // [o][c]
  __shared__ __bf16 Xs[32 * 64];   // [c][n]

  const int n0 = blockIdx.x * 64;
  const int o0 = blockIdx.y * 64;
  const int b  = blockIdx.z;
  const int t  = threadIdx.x;
  const int wave = t >> 5;
  const int lane = t & 31;

  const float* xb = x + (size_t)b * CC * NN;

  v8f acc[4];
#pragma unroll
  for (int i = 0; i < 4; ++i) acc[i] = (v8f){};

  for (int kc = 0; kc < CC; kc += 32) {
    // Stage W tile [64 o][32 c] -> bf16
    {
      const int r  = t >> 1;            // 0..63
      const int c0 = (t & 1) * 16;
      const float* src = W + (size_t)(o0 + r) * CC + kc + c0;
#pragma unroll
      for (int i = 0; i < 16; ++i) Ws[r * 32 + c0 + i] = (__bf16)src[i];
    }
    // Stage x tile [32 c][64 n] -> bf16
    {
      const int r  = t >> 2;            // 0..31
      const int c0 = (t & 3) * 16;
      const float* src = xb + (size_t)(kc + r) * NN + n0 + c0;
#pragma unroll
      for (int i = 0; i < 16; ++i) Xs[r * 64 + c0 + i] = (__bf16)src[i];
    }
    __syncthreads();

    // Wave w computes output rows [o0 + 16w, o0 + 16w + 16)
    const v16bf afrag = load_frag_a(Ws, wave * 16, 0, /*m_stride=*/32, /*k_stride=*/1, lane);
#pragma unroll
    for (int tm = 0; tm < 4; ++tm) {
      const v16bf bfrag = load_frag_b(Xs, 0, tm * 16, /*k_stride=*/64, /*n_stride=*/1, lane);
      acc[tm] = wmma_bf16(afrag, bfrag, acc[tm]);
    }
    __syncthreads();
  }

  // Store C/D layout: col = lane&15, row = vgpr + 8*(lane>=16)
  const int rowoff = (lane >> 4) * 8;
#pragma unroll
  for (int tm = 0; tm < 4; ++tm) {
    const int n = n0 + tm * 16 + (lane & 15);
#pragma unroll
    for (int j = 0; j < 8; ++j) {
      const int o = o0 + wave * 16 + rowoff + j;
      out[(size_t)b * OutC * NN + (size_t)o * NN + n] = (__bf16)acc[tm][j];
    }
  }
}

// ---------------------------------------------------------------------------
// Fused flash attention over the key axis n.
//   S[n,m] = sum_c f[c,n] g[c,m];  beta = softmax_n(S);  O[c,m] = sum_n h[c,n] beta[n,m]
//   out = gamma * O + x
// Workgroup: 256 threads (8 waves). Tile: all 512 c x 64 m. Key blocks of 64.
// Wave w owns O rows c in [64w, 64w+64) (16 accumulator tiles = 128 VGPRs).
// All staging uses global_load_async_to_lds_b128 (ASYNCcnt).
// ---------------------------------------------------------------------------
__global__ __launch_bounds__(256) void attn_kernel(
    const float* __restrict__ x,
    const __bf16* __restrict__ f, const __bf16* __restrict__ g,
    const __bf16* __restrict__ h, const float* __restrict__ gamma,
    float* __restrict__ out) {
  __shared__ __bf16 Qs[64 * 64];    // g tile   [c][m]   8 KB
  __shared__ __bf16 Ks[64 * 64];    // f tile   [c][n]   8 KB
  __shared__ __bf16 Vs[CC * 64];    // h tile   [c][n]  64 KB
  __shared__ float  Ss[64 * 64];    // S scores [m][n]  16 KB  (m-major!)
  __shared__ __bf16 Ps[64 * 64];    // P = exp  [m][n]   8 KB  (m-major!)
  __shared__ float  alphaS[64];     // per-m rescale / inv-sum

  const int m0 = blockIdx.x * 64;
  const int b  = blockIdx.y;
  const int t  = threadIdx.x;
  const int wave = t >> 5;
  const int lane = t & 31;
  const int rowoff = (lane >> 4) * 8;

  const __bf16* fb = f + (size_t)b * C8V * NN;
  const __bf16* gb = g + (size_t)b * C8V * NN;
  const __bf16* hb = h + (size_t)b * CC  * NN;

  // Stage Q tile once: [64 c][64 m], 512 x 16B async transfers
#pragma unroll
  for (int it = 0; it < 2; ++it) {
    const int idx = t + it * 256;
    const int c = idx >> 3, u = idx & 7;
    async_copy_b128((const uint4*)(gb + (size_t)c * NN + m0) + u, ((uint4*)Qs) + idx);
  }

  // S-tile assignment: wave w -> (tn = w&3, tm in {2*(w>>2), 2*(w>>2)+1})
  const int s_tn  = wave & 3;
  const int s_tm0 = (wave >> 2) * 2;

  float m_run = -3.0e38f;   // running max over n, column m = t (threads 0..63)
  float s_run = 0.0f;       // running sum

  v8f accO[4][4];
#pragma unroll
  for (int tc = 0; tc < 4; ++tc)
#pragma unroll
    for (int tm = 0; tm < 4; ++tm) accO[tc][tm] = (v8f){};

  for (int kb = 0; kb < NN / 64; ++kb) {
    const int n0 = kb * 64;

    // --- Async-stage K tile [64 c][64 n] and V tile [512 c][64 n] ---
#pragma unroll
    for (int it = 0; it < 2; ++it) {
      const int idx = t + it * 256;
      const int c = idx >> 3, u = idx & 7;
      async_copy_b128((const uint4*)(fb + (size_t)c * NN + n0) + u, ((uint4*)Ks) + idx);
    }
#pragma unroll
    for (int it = 0; it < 16; ++it) {
      const int idx = t + it * 256;
      const int c = idx >> 3, u = idx & 7;
      async_copy_b128((const uint4*)(hb + (size_t)c * NN + n0) + u, ((uint4*)Vs) + idx);
    }
    wait_async0();
    __syncthreads();

    // --- S = K^T Q : each wave computes 2 of the 16 16x16 tiles ---
    {
      v8f sacc[2];
      sacc[0] = (v8f){};
      sacc[1] = (v8f){};
#pragma unroll
      for (int ks = 0; ks < 2; ++ks) {
        // A[M=n][K=c] from Ks stored [c][n] -> addr = c*64 + n
        const v16bf af = load_frag_a(Ks, s_tn * 16, ks * 32, /*m_stride=*/1, /*k_stride=*/64, lane);
#pragma unroll
        for (int q = 0; q < 2; ++q) {
          const v16bf bq = load_frag_b(Qs, ks * 32, (s_tm0 + q) * 16, 64, 1, lane);
          sacc[q] = wmma_bf16(af, bq, sacc[q]);
        }
      }
      // Write S m-major: lane column m fixed, 8 consecutive n per lane
#pragma unroll
      for (int q = 0; q < 2; ++q) {
        const int m = (s_tm0 + q) * 16 + (lane & 15);
        const int nbase = s_tn * 16 + rowoff;
#pragma unroll
        for (int j = 0; j < 8; ++j) {
          Ss[m * 64 + nbase + j] = sacc[q][j];
        }
      }
    }
    __syncthreads();

    // --- Online softmax over n: thread t<64 owns column m=t (contiguous row) ---
    if (t < 64) {
      float bm = -3.0e38f;
#pragma unroll 8
      for (int n = 0; n < 64; ++n) bm = fmaxf(bm, Ss[t * 64 + n]);
      const float nm = fmaxf(m_run, bm);
      const float al = __expf(m_run - nm);   // == 0 on first block (underflow)
      float ssum = 0.0f;
#pragma unroll 8
      for (int n = 0; n < 64; ++n) {
        const float p = __expf(Ss[t * 64 + n] - nm);
        Ps[t * 64 + n] = (__bf16)p;          // P stored bf16, m-major
        ssum += p;
      }
      s_run = s_run * al + ssum;
      m_run = nm;
      alphaS[t] = al;
    }
    __syncthreads();

    // --- Rescale O accumulators by alpha[m] ---
#pragma unroll
    for (int tm = 0; tm < 4; ++tm) {
      const float al = alphaS[tm * 16 + (lane & 15)];
#pragma unroll
      for (int tc = 0; tc < 4; ++tc)
#pragma unroll
        for (int j = 0; j < 8; ++j) accO[tc][tm][j] *= al;
    }

    // --- O += V * P : wave w handles c rows [64w, 64w+64) ---
#pragma unroll
    for (int ks = 0; ks < 2; ++ks) {
      v16bf pb[4];
#pragma unroll
      for (int tm = 0; tm < 4; ++tm)
        // B[K=n][N=m] from Ps stored [m][n] -> addr = m*64 + n (K contiguous)
        pb[tm] = load_frag_b(Ps, ks * 32, tm * 16, /*k_stride=*/1, /*n_stride=*/64, lane);
#pragma unroll
      for (int tc = 0; tc < 4; ++tc) {
        // A[M=c][K=n] from Vs stored [c][n] -> addr = c*64 + n (K contiguous)
        const v16bf av = load_frag_a(Vs, wave * 64 + tc * 16, ks * 32, 64, 1, lane);
#pragma unroll
        for (int tm = 0; tm < 4; ++tm)
          accO[tc][tm] = wmma_bf16(av, pb[tm], accO[tc][tm]);
      }
    }
    __syncthreads();
  }

  // --- Finalize: out = gamma * O / sum + x ---
  if (t < 64) alphaS[t] = 1.0f / s_run;
  __syncthreads();

  const float gm = gamma[0];
  const float* xb = x   + (size_t)b * CC * NN;
  float*       ob = out + (size_t)b * CC * NN;
#pragma unroll
  for (int tm = 0; tm < 4; ++tm) {
    const int mloc = tm * 16 + (lane & 15);
    const int m = m0 + mloc;
    const float inv = alphaS[mloc];
#pragma unroll
    for (int tc = 0; tc < 4; ++tc) {
#pragma unroll
      for (int j = 0; j < 8; ++j) {
        const int c = wave * 64 + tc * 16 + rowoff + j;
        ob[(size_t)c * NN + m] = gm * accO[tc][tm][j] * inv + xb[(size_t)c * NN + m];
      }
    }
  }
}

// ---------------------------------------------------------------------------
extern "C" void kernel_launch(void* const* d_in, const int* in_sizes, int n_in,
                              void* d_out, int out_size, void* d_ws, size_t ws_size,
                              hipStream_t stream) {
  const float* x     = (const float*)d_in[0];
  const float* Wq    = (const float*)d_in[1];
  const float* Wk    = (const float*)d_in[2];
  const float* Wv    = (const float*)d_in[3];
  const float* gamma = (const float*)d_in[4];
  float* out = (float*)d_out;

  // Workspace (bf16): f [B,64,N] (2MB), g [B,64,N] (2MB), h [B,512,N] (16MB)
  __bf16* fbuf = (__bf16*)d_ws;
  __bf16* gbuf = fbuf + (size_t)BB * C8V * NN;
  __bf16* hbuf = gbuf + (size_t)BB * C8V * NN;

  proj_kernel<<<dim3(NN / 64, C8V / 64, BB), 128, 0, stream>>>(x, Wq, fbuf, C8V);
  proj_kernel<<<dim3(NN / 64, C8V / 64, BB), 128, 0, stream>>>(x, Wk, gbuf, C8V);
  proj_kernel<<<dim3(NN / 64, CC  / 64, BB), 128, 0, stream>>>(x, Wv, hbuf, CC);
  attn_kernel<<<dim3(NN / 64, BB), 256, 0, stream>>>(x, fbuf, gbuf, hbuf, gamma, out);
}
